// MultiHeadAttention_47751446397550
// MI455X (gfx1250) — matrix operand order repaired and checked by reading, not compiled, not run on hardware
//
#include <hip/hip_runtime.h>
#include <hip/hip_bf16.h>

typedef __attribute__((ext_vector_type(16))) __bf16 v16bf;
typedef __attribute__((ext_vector_type(8)))  float  v8f;
typedef __attribute__((ext_vector_type(4)))  unsigned int u32x4;
typedef __attribute__((ext_vector_type(8)))  int i32x8;
typedef __attribute__((ext_vector_type(4)))  int i32x4;

#define BATCH  2
#define SEQ    2048
#define DMODEL 1024
#define NHEADS 16
#define HDIM   64
#define MTOT   (BATCH * SEQ)                         // 4096 tokens
#define PLANE  ((size_t)BATCH * NHEADS * SEQ * HDIM) // elems per Q/K/V plane

#if defined(__HIP_DEVICE_COMPILE__) && \
    __has_builtin(__builtin_amdgcn_tensor_load_to_lds) && \
    __has_builtin(__builtin_amdgcn_s_wait_tensorcnt)
#define USE_TDM 1
#else
#define USE_TDM 0
#endif

union F16x16 { uint4 u[2]; v16bf v; };
union B8     { uint4 u;    __bf16 h[8]; };

// round-to-nearest-even f32 -> bf16
__device__ __forceinline__ __bf16 cvt_bf16(float f) {
  unsigned int u = __builtin_bit_cast(unsigned int, f);
  u += 0x7fffu + ((u >> 16) & 1u);
  unsigned short s = (unsigned short)(u >> 16);
  return __builtin_bit_cast(__bf16, s);
}

// A-fragment (16x32 bf16): K contiguous per half -> one 32B run per lane
__device__ __forceinline__ v16bf load_afrag(const __bf16* row, int hi) {
  F16x16 f;
  f.u[0] = *(const uint4*)(row + hi * 16);
  f.u[1] = *(const uint4*)(row + hi * 16 + 8);
  return f.v;
}
// B-fragment: lane = output column, K contiguous per half -> one 32B run
__device__ __forceinline__ v16bf load_bfrag(const __bf16* p) {
  F16x16 f;
  f.u[0] = *(const uint4*)p;
  f.u[1] = *(const uint4*)(p + 8);
  return f.v;
}

#if USE_TDM
// D# group1 word0: data_size=2B + LDS pad config matching our padded strides
#define W0_PAD64  ((1u << 16) | (1u << 20) | (3u << 22) | (3u << 25)) // 64B rows, +16B pad
#define W0_PAD128 ((1u << 16) | (1u << 20) | (4u << 22) | (3u << 25)) // 128B rows, +16B pad

__device__ __forceinline__ unsigned lds_off(const void* p) {
  return (unsigned)(size_t)p;   // low 32 bits of generic LDS address = byte offset
}

// Issue a 2D TDM tile load: tile [tile_d1 rows x tile_d0 elems] of bf16 from a
// row-major tensor [tensor_d1 x tensor_d0] (row stride stride0 elems) into LDS.
__device__ __forceinline__ void tdm_load_2d(const void* gsrc, unsigned ldsaddr,
                                            unsigned w0,
                                            unsigned tensor_d0, unsigned tensor_d1,
                                            unsigned tile_d0, unsigned tile_d1,
                                            unsigned stride0) {
  unsigned long long ga = (unsigned long long)(size_t)gsrc;
  u32x4 g0;
  g0[0] = 1u;                                              // count=1, user D#
  g0[1] = ldsaddr;                                         // lds_addr (bytes)
  g0[2] = (unsigned)ga;                                    // global_addr[31:0]
  g0[3] = (unsigned)((ga >> 32) & 0x1FFFFFFu) | (2u << 30);// addr[56:32] | type=2
  i32x8 g1;
  g1[0] = (int)w0;
  g1[1] = (int)(tensor_d0 << 16);                          // tensor_dim0 lo
  g1[2] = (int)((tensor_d0 >> 16) | (tensor_d1 << 16));    // dim0 hi | dim1 lo
  g1[3] = (int)((tensor_d1 >> 16) | (tile_d0 << 16));      // dim1 hi | tile_dim0
  g1[4] = (int)tile_d1;                                    // tile_dim1 (tile_dim2=0)
  g1[5] = (int)stride0;                                    // tensor_dim0_stride lo
  g1[6] = 0; g1[7] = 0;
  i32x4 z4 = {0, 0, 0, 0};
#if __clang_major__ >= 23
  i32x8 z8 = {0, 0, 0, 0, 0, 0, 0, 0};
  __builtin_amdgcn_tensor_load_to_lds(g0, g1, z4, z4, z8, 0);
#else
  __builtin_amdgcn_tensor_load_to_lds(g0, g1, z4, z4, 0);
#endif
}
#endif // USE_TDM

// ---------------------------------------------------------------------------
// Prepack: f32 -> bf16 (vectorized)
// ---------------------------------------------------------------------------
__global__ __launch_bounds__(256)
void pack_bf16_kernel(const float* __restrict__ src, __bf16* __restrict__ dst, int n) {
  int i = (blockIdx.x * 256 + threadIdx.x) * 8;
  if (i >= n) return;
  float4 f0 = *(const float4*)(src + i);
  float4 f1 = *(const float4*)(src + i + 4);
  B8 b;
  b.h[0] = cvt_bf16(f0.x); b.h[1] = cvt_bf16(f0.y);
  b.h[2] = cvt_bf16(f0.z); b.h[3] = cvt_bf16(f0.w);
  b.h[4] = cvt_bf16(f1.x); b.h[5] = cvt_bf16(f1.y);
  b.h[6] = cvt_bf16(f1.z); b.h[7] = cvt_bf16(f1.w);
  *(uint4*)(dst + i) = b.u;
}

// ---------------------------------------------------------------------------
// Prepack: W[k][n] f32 -> Wt[n][k] bf16 (LDS tile transpose); z selects matrix
// ---------------------------------------------------------------------------
__global__ __launch_bounds__(256)
void transpose_pack_kernel(const float* __restrict__ W0, const float* __restrict__ W1,
                           const float* __restrict__ W2, const float* __restrict__ W3,
                           __bf16* __restrict__ wt) {
  __shared__ __bf16 t[32][33];
  const float* W = (blockIdx.z == 0) ? W0 : (blockIdx.z == 1) ? W1
                 : (blockIdx.z == 2) ? W2 : W3;
  __bf16* Wt = wt + (size_t)blockIdx.z * DMODEL * DMODEL;
  const int tx = threadIdx.x, ty = threadIdx.y;      // 32 x 8
  const int kb = blockIdx.y * 32, nb = blockIdx.x * 32;
  #pragma unroll
  for (int r = 0; r < 4; ++r)
    t[ty + r * 8][tx] = cvt_bf16(W[(kb + ty + r * 8) * DMODEL + nb + tx]);
  __syncthreads();
  #pragma unroll
  for (int r = 0; r < 4; ++r)
    Wt[(nb + ty + r * 8) * DMODEL + kb + tx] = t[tx][ty + r * 8];
}

// ---------------------------------------------------------------------------
// 128x128-tile bf16 WMMA GEMM core. 256 threads (8 waves, 4x2 grid of 32x64
// wave tiles). Double-buffered LDS; staging by the Tensor Data Mover.
// ---------------------------------------------------------------------------
__device__ __forceinline__ void gemm128(const __bf16* __restrict__ A,
                                        const __bf16* __restrict__ Wt,
                                        int mbase, int nbase, int arows,
                                        __bf16 (*xs)[128 * 40], __bf16 (*ws)[128 * 40],
                                        v8f acc[2][4]) {
  const int tid  = threadIdx.x;
  const int lane = tid & 31;
  const int wid  = tid >> 5;
  const int wr   = wid & 3;           // wave row group: 32 rows each
  const int wc   = wid >> 2;          // wave col group: 64 cols each
  const int n0   = lane & 15;
  const int hi   = lane >> 4;
  const v8f vzero = {0.f,0.f,0.f,0.f,0.f,0.f,0.f,0.f};
  #pragma unroll
  for (int i = 0; i < 2; ++i)
    #pragma unroll
    for (int nt = 0; nt < 4; ++nt) acc[i][nt] = vzero;

#if USE_TDM
  // wave-uniform issuer flag -> scalar branch, no exec juggling
  const bool issuer = ((__builtin_amdgcn_readfirstlane((int)threadIdx.x) >> 5) == 0);
  if (issuer) {
    tdm_load_2d(A  + (size_t)mbase * DMODEL, lds_off(xs[0]), W0_PAD64,
                DMODEL, (unsigned)arows, 32, 128, DMODEL);
    tdm_load_2d(Wt + (size_t)nbase * DMODEL, lds_off(ws[0]), W0_PAD64,
                DMODEL, DMODEL, 32, 128, DMODEL);
  }
#endif

  for (int kb = 0, ib = 0; kb < DMODEL; kb += 32, ib ^= 1) {
#if USE_TDM
    if (issuer) __builtin_amdgcn_s_wait_tensorcnt(0);     // current tile landed
    __syncthreads();                                       // publish to all waves
    if (issuer && kb + 32 < DMODEL) {                      // overlap next DMA
      tdm_load_2d(A  + (size_t)mbase * DMODEL + kb + 32, lds_off(xs[ib ^ 1]),
                  W0_PAD64, DMODEL, (unsigned)arows, 32, 128, DMODEL);
      tdm_load_2d(Wt + (size_t)nbase * DMODEL + kb + 32, lds_off(ws[ib ^ 1]),
                  W0_PAD64, DMODEL, DMODEL, 32, 128, DMODEL);
    }
#else
    #pragma unroll
    for (int j = 0; j < 2; ++j) {
      int c = tid + j * 256;
      int r = c >> 2, q = c & 3;
      *(uint4*)(xs[ib] + r * 40 + q * 8) =
          *(const uint4*)(A + (size_t)(mbase + r) * DMODEL + kb + q * 8);
      *(uint4*)(ws[ib] + r * 40 + q * 8) =
          *(const uint4*)(Wt + (size_t)(nbase + r) * DMODEL + kb + q * 8);
    }
    __syncthreads();
#endif

    v16bf af[2];
    #pragma unroll
    for (int i = 0; i < 2; ++i)
      af[i] = load_afrag(xs[ib] + (wr * 32 + i * 16 + n0) * 40, hi);
    #pragma unroll
    for (int nt = 0; nt < 4; ++nt) {
      v16bf bf = load_bfrag(ws[ib] + (wc * 64 + nt * 16 + n0) * 40 + hi * 16);
      #pragma unroll
      for (int i = 0; i < 2; ++i)
        acc[i][nt] = __builtin_amdgcn_wmma_f32_16x16x32_bf16(
            false, af[i], false, bf, (short)0, acc[i][nt], false, false);
    }
    __syncthreads();
  }
}

// ---------------------------------------------------------------------------
// Kernel 1: Q/K/V projections. Q,K stored [b,h,t,d]; V stored transposed [b,h,d,t]
// ---------------------------------------------------------------------------
__global__ __launch_bounds__(256)
void qkv_proj_kernel(const __bf16* __restrict__ xbf,
                     const __bf16* __restrict__ wt,
                     __bf16* __restrict__ qkv) {
  __shared__ alignas(16) __bf16 xs[2][128 * 40];
  __shared__ alignas(16) __bf16 ws[2][128 * 40];
  const int mbase = blockIdx.x * 128;
  const int nbase = blockIdx.y * 128;
  const __bf16* Wt = wt + (size_t)blockIdx.z * DMODEL * DMODEL;
  __bf16* out = qkv + (size_t)blockIdx.z * PLANE;

  v8f acc[2][4];
  gemm128(xbf, Wt, mbase, nbase, MTOT, xs, ws, acc);

  const int lane = threadIdx.x & 31;
  const int wid  = threadIdx.x >> 5;
  const int wr = wid & 3, wc = wid >> 2;
  const int n0 = lane & 15, hi = lane >> 4;
  const bool isV = (blockIdx.z == 2);
  #pragma unroll
  for (int i = 0; i < 2; ++i)
    #pragma unroll
    for (int nt = 0; nt < 4; ++nt)
      #pragma unroll
      for (int v = 0; v < 8; ++v) {
        int row = mbase + wr * 32 + i * 16 + v + hi * 8;   // token
        int col = nbase + wc * 64 + nt * 16 + n0;          // feature
        int b = row >> 11, t = row & (SEQ - 1);
        int h = col >> 6,  d = col & (HDIM - 1);
        size_t base = (size_t)(b * NHEADS + h) * SEQ * HDIM;
        __bf16 val = cvt_bf16(acc[i][nt][v]);
        if (isV) out[base + (size_t)d * SEQ + t] = val;    // [b,h,d,t]
        else     out[base + (size_t)t * HDIM + d] = val;   // [b,h,t,d]
      }
}

// ---------------------------------------------------------------------------
// Kernel 2: flash-style causal attention. 1 wave per 32-query tile per (b,h):
// each K/V tile is amortized over 32 queries (2 row fragments, 16 WMMAs/tile).
// K/V tiles streamed by TDM with double buffering.
// ---------------------------------------------------------------------------
__global__ __launch_bounds__(32)
void attn_kernel(const __bf16* __restrict__ qkv, __bf16* __restrict__ ctxout) {
  __shared__ alignas(16) __bf16 kl[2][32 * 72];  // keys x dims, stride 72 bf16
  __shared__ alignas(16) __bf16 vt[2][64 * 40];  // dims x keys, stride 40 bf16
  __shared__ alignas(16) __bf16 pl[32 * 40];     // P 32x32, stride 40

  const int lane = threadIdx.x;
  const int qt = blockIdx.x, h = blockIdx.y, b = blockIdx.z;
  const int qbase = qt * 32;
  const size_t bh = ((size_t)b * NHEADS + h) * SEQ * HDIM;
  const __bf16* Q  = qkv + bh;                   // [t][d]
  const __bf16* K  = qkv + PLANE + bh;           // [t][d]
  const __bf16* Vt = qkv + 2 * PLANE + bh;       // [d][t]

  const int n0 = lane & 15;
  const int hi = lane >> 4;

  // Q fragments: 2 row tiles x 2 K-steps
  v16bf qa[2][2];
  #pragma unroll
  for (int r = 0; r < 2; ++r)
    #pragma unroll
    for (int s = 0; s < 2; ++s)
      qa[r][s] = load_afrag(Q + (qbase + r * 16 + n0) * HDIM + s * 32, hi);

  const v8f vzero = {0.f,0.f,0.f,0.f,0.f,0.f,0.f,0.f};
  v8f ctx[2][4];
  float mi[2][8], li[2][8];
  #pragma unroll
  for (int r = 0; r < 2; ++r) {
    #pragma unroll
    for (int nt = 0; nt < 4; ++nt) ctx[r][nt] = vzero;
    #pragma unroll
    for (int v = 0; v < 8; ++v) { mi[r][v] = -1e30f; li[r][v] = 0.f; }
  }

  const int nk = qbase / 32 + 1;   // number of 32-key tiles (causal)

#if USE_TDM
  tdm_load_2d(K, lds_off(kl[0]), W0_PAD128, HDIM, SEQ, HDIM, 32, HDIM);
  tdm_load_2d(Vt, lds_off(vt[0]), W0_PAD64, SEQ, HDIM, 32, HDIM, SEQ);
#endif

  for (int it = 0; it < nk; ++it) {
    const int ib = it & 1;
    const int kbase = it * 32;
#if USE_TDM
    if (it + 1 < nk) {
      __syncthreads();   // drain ds reads of buffer ib^1 before DMA reuse
      const int kn = (it + 1) * 32;
      tdm_load_2d(K + (size_t)kn * HDIM, lds_off(kl[ib ^ 1]),
                  W0_PAD128, HDIM, SEQ, HDIM, 32, HDIM);
      tdm_load_2d(Vt + kn, lds_off(vt[ib ^ 1]),
                  W0_PAD64, SEQ, HDIM, 32, HDIM, SEQ);
      __builtin_amdgcn_s_wait_tensorcnt(2);   // current tile done, next in flight
    } else {
      __builtin_amdgcn_s_wait_tensorcnt(0);
    }
#else
    __syncthreads();
    {
      const uint4* gk = (const uint4*)(K + (size_t)(kbase + lane) * HDIM);
      uint4* dk = (uint4*)(kl[ib] + lane * 72);
      #pragma unroll
      for (int i = 0; i < 8; ++i) dk[i] = gk[i];
      #pragma unroll
      for (int i = 0; i < 8; ++i) {
        int c = lane + i * 32;
        int r = c >> 2, j = c & 3;
        *(uint4*)(vt[ib] + r * 40 + j * 8) =
            *(const uint4*)(Vt + (size_t)r * SEQ + kbase + j * 8);
      }
      if (kbase + 32 <= qbase + 31)
        __builtin_prefetch(K + (size_t)(kbase + 32 + lane) * HDIM, 0, 0);
    }
    __syncthreads();
#endif

    // S = Q K^T : 2 row tiles x 2 key blocks, fp32 accumulate (8 WMMA)
    v8f s[2][2];
    #pragma unroll
    for (int r = 0; r < 2; ++r) { s[r][0] = vzero; s[r][1] = vzero; }
    #pragma unroll
    for (int ks = 0; ks < 2; ++ks) {
      v16bf bk0 = load_bfrag(kl[ib] + n0        * 72 + ks * 32 + hi * 16);
      v16bf bk1 = load_bfrag(kl[ib] + (16 + n0) * 72 + ks * 32 + hi * 16);
      #pragma unroll
      for (int r = 0; r < 2; ++r) {
        s[r][0] = __builtin_amdgcn_wmma_f32_16x16x32_bf16(false, qa[r][ks], false, bk0, (short)0, s[r][0], false, false);
        s[r][1] = __builtin_amdgcn_wmma_f32_16x16x32_bf16(false, qa[r][ks], false, bk1, (short)0, s[r][1], false, false);
      }
    }

    // per-row-tile causal mask + scale + online softmax
    const float sc = 0.125f;
    #pragma unroll
    for (int r = 0; r < 2; ++r) {
      float p0[8], p1[8], rmax[8];
      #pragma unroll
      for (int v = 0; v < 8; ++v) {
        int qrow = qbase + r * 16 + v + hi * 8;
        float a0 = (kbase + n0      <= qrow) ? s[r][0][v] * sc : -1e30f;
        float a1 = (kbase + 16 + n0 <= qrow) ? s[r][1][v] * sc : -1e30f;
        p0[v] = a0; p1[v] = a1;
        rmax[v] = fmaxf(a0, a1);
      }
      #pragma unroll
      for (int off = 1; off < 16; off <<= 1)   // stays within 16-lane halves
        #pragma unroll
        for (int v = 0; v < 8; ++v)
          rmax[v] = fmaxf(rmax[v], __shfl_xor(rmax[v], off, 32));

      float rsum[8];
      #pragma unroll
      for (int v = 0; v < 8; ++v) {
        float mn = fmaxf(mi[r][v], rmax[v]);
        float f  = __expf(mi[r][v] - mn);
        mi[r][v] = mn;
        float e0 = __expf(p0[v] - mn);
        float e1 = __expf(p1[v] - mn);
        pl[(r * 16 + v + hi * 8) * 40 + n0]      = cvt_bf16(e0);
        pl[(r * 16 + v + hi * 8) * 40 + 16 + n0] = cvt_bf16(e1);
        rsum[v] = e0 + e1;
        li[r][v] *= f;
        #pragma unroll
        for (int nt = 0; nt < 4; ++nt) ctx[r][nt][v] *= f;
      }
      #pragma unroll
      for (int off = 1; off < 16; off <<= 1)
        #pragma unroll
        for (int v = 0; v < 8; ++v)
          rsum[v] += __shfl_xor(rsum[v], off, 32);
      #pragma unroll
      for (int v = 0; v < 8; ++v)
        li[r][v] += rsum[v];
    }

    // ctx += P @ V (same-wave DS ops are in-order: P store -> A-frag load safe)
    v16bf pa[2];
    #pragma unroll
    for (int r = 0; r < 2; ++r)
      pa[r] = load_afrag(pl + (r * 16 + n0) * 40, hi);
    #pragma unroll
    for (int nt = 0; nt < 4; ++nt) {
      v16bf bv = load_bfrag(vt[ib] + (nt * 16 + n0) * 40 + hi * 16);
      #pragma unroll
      for (int r = 0; r < 2; ++r)
        ctx[r][nt] = __builtin_amdgcn_wmma_f32_16x16x32_bf16(
            false, pa[r], false, bv, (short)0, ctx[r][nt], false, false);
    }
  }

  // normalize + store bf16 context merged to [b, t, h*64+d]
  #pragma unroll
  for (int r = 0; r < 2; ++r)
    #pragma unroll
    for (int v = 0; v < 8; ++v) {
      float inv = 1.0f / li[r][v];
      int row = b * SEQ + qbase + r * 16 + v + hi * 8;
      #pragma unroll
      for (int nt = 0; nt < 4; ++nt) {
        int col = h * HDIM + nt * 16 + n0;
        ctxout[(size_t)row * DMODEL + col] = cvt_bf16(ctx[r][nt][v] * inv);
      }
    }
}

// ---------------------------------------------------------------------------
// Kernel 3: out = ctx @ Wo + bo   (fp32 out)
// ---------------------------------------------------------------------------
__global__ __launch_bounds__(256)
void out_proj_kernel(const __bf16* __restrict__ ctx,
                     const __bf16* __restrict__ Wot,
                     const float* __restrict__ bo,
                     float* __restrict__ out) {
  __shared__ alignas(16) __bf16 xs[2][128 * 40];
  __shared__ alignas(16) __bf16 ws[2][128 * 40];
  const int mbase = blockIdx.x * 128;
  const int nbase = blockIdx.y * 128;
  v8f acc[2][4];
  gemm128(ctx, Wot, mbase, nbase, MTOT, xs, ws, acc);

  const int lane = threadIdx.x & 31;
  const int wid  = threadIdx.x >> 5;
  const int wr = wid & 3, wc = wid >> 2;
  const int n0 = lane & 15, hi = lane >> 4;
  #pragma unroll
  for (int i = 0; i < 2; ++i)
    #pragma unroll
    for (int nt = 0; nt < 4; ++nt) {
      int col = nbase + wc * 64 + nt * 16 + n0;
      float bias = bo[col];
      #pragma unroll
      for (int v = 0; v < 8; ++v) {
        int row = mbase + wr * 32 + i * 16 + v + hi * 8;
        out[(size_t)row * DMODEL + col] = acc[i][nt][v] + bias;
      }
    }
}

// ---------------------------------------------------------------------------
extern "C" void kernel_launch(void* const* d_in, const int* in_sizes, int n_in,
                              void* d_out, int out_size, void* d_ws, size_t ws_size,
                              hipStream_t stream) {
  (void)in_sizes; (void)n_in; (void)out_size; (void)ws_size;
  const float* x  = (const float*)d_in[0];
  const float* Wq = (const float*)d_in[1];
  const float* Wk = (const float*)d_in[2];
  const float* Wv = (const float*)d_in[3];
  const float* Wo = (const float*)d_in[4];
  const float* bo = (const float*)d_in[5];
  float* out = (float*)d_out;

  // workspace: [x bf16][4x Wt bf16][Q|K|Vt bf16][ctx bf16]
  char* p = (char*)d_ws;
  __bf16* xbf  = (__bf16*)p;                 p += (size_t)MTOT * DMODEL * 2;
  __bf16* wt   = (__bf16*)p;                 p += 4ull * DMODEL * DMODEL * 2;
  __bf16* qkv  = (__bf16*)p;                 p += 3ull * PLANE * 2;
  __bf16* ctxb = (__bf16*)p;

  pack_bf16_kernel<<<dim3((MTOT * DMODEL / 8) / 256), 256, 0, stream>>>(x, xbf, MTOT * DMODEL);
  transpose_pack_kernel<<<dim3(32, 32, 4), dim3(32, 8), 0, stream>>>(Wq, Wk, Wv, Wo, wt);
  qkv_proj_kernel<<<dim3(MTOT / 128, DMODEL / 128, 3), 256, 0, stream>>>(xbf, wt, qkv);
  attn_kernel<<<dim3(SEQ / 32, NHEADS, BATCH), 32, 0, stream>>>(qkv, ctxb);
  out_proj_kernel<<<dim3(MTOT / 128, DMODEL / 128, 1), 256, 0, stream>>>(
      ctxb, wt + 3ull * DMODEL * DMODEL, bo, out);
}
